// Net_51015621542483
// MI455X (gfx1250) — matrix operand order repaired and checked by reading, not compiled, not run on hardware
//
#include <hip/hip_runtime.h>
#include <hip/hip_bf16.h>
#include <math.h>

// ---------------------------------------------------------------------------
// Net: emb gather -> fc1(relu) -> 512-step RNN scan (sigmoid, linear, max-pool)
//      -> fc2.  bf16 WMMA (v_wmma_f32_16x16x32_bf16) everywhere, f32 accum.
// Scan kernel async-prefetches next timestep's activations into LDS with
// global_load_async_to_lds_b128 (ASYNCcnt) while computing the current step.
// ---------------------------------------------------------------------------

typedef __attribute__((ext_vector_type(8)))  float  v8f;
typedef __attribute__((ext_vector_type(8)))  __bf16 v8bf;
typedef __attribute__((ext_vector_type(16))) __bf16 v16bf;

#define NB   64      // batch
#define TT   512     // timesteps
#define NE   300     // emb dim
#define NH   300     // hidden dim
#define NP   304     // N padded to 16
#define KE   320     // K padded to 32 (fc1 / l2)
#define KC   608     // K padded concat (600 -> 608, with 4-pad at 300..303)
#define HS   320     // hidden row stride (bf16 elems)
#define NWG  19      // persistent workgroups in the scan (= NP/16)

static __device__ inline v8f vzero8() {
  v8f z;
#pragma unroll
  for (int i = 0; i < 8; ++i) z[i] = 0.f;
  return z;
}
static __device__ inline v8f vneginf8() {
  v8f z;
#pragma unroll
  for (int i = 0; i < 8; ++i) z[i] = -INFINITY;
  return z;
}
static __device__ inline v16bf cat16(v8bf lo, v8bf hi) {
  return __builtin_shufflevector(lo, hi, 0,1,2,3,4,5,6,7,8,9,10,11,12,13,14,15);
}
// Unguarded 8-wide f32 -> bf16 run (two 16B loads, fully in-bounds).
static __device__ inline v8bf load8_f32_bf_nc(const float* __restrict__ base, int kb) {
  const float4 f0 = *(const float4*)(base + kb);
  const float4 f1 = *(const float4*)(base + kb + 4);
  v8bf r;
  r[0] = (__bf16)f0.x; r[1] = (__bf16)f0.y; r[2] = (__bf16)f0.z; r[3] = (__bf16)f0.w;
  r[4] = (__bf16)f1.x; r[5] = (__bf16)f1.y; r[6] = (__bf16)f1.z; r[7] = (__bf16)f1.w;
  return r;
}
// Guarded tail variant (only used for the final K iteration).
static __device__ inline v8bf load8_f32_bf(const float* __restrict__ base, int kb, int kmax) {
  v8bf r;
#pragma unroll
  for (int i = 0; i < 8; ++i) r[i] = (__bf16)((kb + i < kmax) ? base[kb + i] : 0.f);
  return r;
}

// Device-wide sense barrier for the persistent scan kernel (19 resident WGs).
static __device__ inline void gbar(int* cnt, int target) {
  __threadfence();
  __syncthreads();
  if (threadIdx.x == 0) {
    __atomic_fetch_add(cnt, 1, __ATOMIC_ACQ_REL);
    while (__atomic_load_n(cnt, __ATOMIC_ACQUIRE) < target) {
      __builtin_amdgcn_s_sleep(1);
    }
  }
  __syncthreads();
  __threadfence();
}

// Issue async global->LDS copy of one timestep tile (NB*NP bf16 = 38,912 B)
// in 16B granules, spread over the 128 threads (19 granules each).
static __device__ inline void async_copy_xt(const __bf16* __restrict__ src,
                                            unsigned lds_byte_base, int tid) {
#pragma unroll 1
  for (int g = tid; g < (NB * NP * 2) / 16; g += 128) {
    unsigned long long ga = (unsigned long long)(uintptr_t)(src) +
                            (unsigned long long)g * 16ull;
    unsigned la = lds_byte_base + (unsigned)g * 16u;
    asm volatile("global_load_async_to_lds_b128 %0, %1, off"
                 :: "v"(la), "v"(ga) : "memory");
  }
}
static __device__ inline void wait_async0() {
  asm volatile("s_wait_asynccnt 0x0" ::: "memory");
}

// ---------------------------------------------------------------------------
// Kernel 0: pack weights into bf16 B-operand layout ([K][N] transposed, padded
// with zeros), pad biases, zero the hidden ping-pong buffers + barrier counter.
// ---------------------------------------------------------------------------
__global__ void pack_kernel(const float* __restrict__ fc1_w, const float* __restrict__ fc1_b,
                            const float* __restrict__ l1_w,  const float* __restrict__ l1_b,
                            const float* __restrict__ l2_w,  const float* __restrict__ l2_b,
                            __bf16* fc1p, __bf16* b1p, __bf16* b2p,
                            float* fc1_bp, float* l1_bp, float* l2_bp,
                            __bf16* h0, __bf16* h1, int* cnt) {
  const int S0 = KE * NP, S1 = KC * NP, S2 = KE * NP;
  const int S3 = NP, S4 = NP, S5 = NP, S6 = NB * HS, S7 = NB * HS, S8 = 16;
  const int total = S0 + S1 + S2 + S3 + S4 + S5 + S6 + S7 + S8;
  for (int i = blockIdx.x * blockDim.x + threadIdx.x; i < total;
       i += gridDim.x * blockDim.x) {
    int j = i;
    if (j < S0) {                       // fc1p[k][n] = fc1_w[n][k]
      int k = j / NP, n = j % NP;
      fc1p[j] = (__bf16)((k < NE && n < NE) ? fc1_w[n * NE + k] : 0.f);
      continue;
    }
    j -= S0;
    if (j < S1) {                       // b1p[k'][n] = l1_w[n][map(k')]
      int k = j / NP, n = j % NP;
      int kk = (k < 300) ? k : ((k >= 304 && k < 604) ? (k - 4) : -1);
      b1p[j] = (__bf16)((kk >= 0 && n < NH) ? l1_w[n * 600 + kk] : 0.f);
      continue;
    }
    j -= S1;
    if (j < S2) {                       // b2p[k][n] = l2_w[n][k]
      int k = j / NP, n = j % NP;
      b2p[j] = (__bf16)((k < NH && n < NH) ? l2_w[n * NH + k] : 0.f);
      continue;
    }
    j -= S2;
    if (j < S3) { fc1_bp[j] = (j < NE) ? fc1_b[j] : 0.f; continue; }
    j -= S3;
    if (j < S4) { l1_bp[j] = (j < NH) ? l1_b[j] : 0.f; continue; }
    j -= S4;
    if (j < S5) { l2_bp[j] = (j < NH) ? l2_b[j] : 0.f; continue; }
    j -= S5;
    if (j < S6) { h0[j] = (__bf16)0.f; continue; }
    j -= S6;
    if (j < S7) { h1[j] = (__bf16)0.f; continue; }
    j -= S7;
    cnt[j] = 0;
  }
}

// ---------------------------------------------------------------------------
// Kernel 1: Hx[t*64+b][0..303] = relu(emb[x[b][t]] @ fc1_w^T + fc1_b) in bf16.
// One wave per 16x16 output tile; 9 unguarded K iterations + 1 guarded tail.
// ---------------------------------------------------------------------------
__global__ __launch_bounds__(128) void embed_fc1_kernel(
    const int* __restrict__ x, const float* __restrict__ emb,
    const __bf16* __restrict__ fc1p, const float* __restrict__ fc1_bp,
    __bf16* __restrict__ Hx) {
  const int tid = threadIdx.x;
  const int wave = tid >> 5, lane = tid & 31;
  const int half = lane >> 4, lcol = lane & 15;
  const int tile = blockIdx.x * 4 + wave;          // 2048*19 tiles
  const int m0 = (tile / NWG) * 16, n0 = (tile % NWG) * 16;

  const int am = m0 + lcol;                        // global row = t*64 + b
  const int t = am >> 6, b = am & 63;
  const int vocab = x[b * TT + t];
  const float* erow = emb + (size_t)vocab * NE;

  v8f c = vzero8();
  // main loop: K = 0..287, all 8-runs fully inside [0,300)
#pragma unroll 3
  for (int kk = 0; kk < 9; ++kk) {
    const int k0 = kk * 32;
    v16bf a = cat16(load8_f32_bf_nc(erow, k0 + 8 * half),
                    load8_f32_bf_nc(erow, k0 + 16 + 8 * half));
    v16bf bm;
#pragma unroll
    for (int s = 0; s < 16; ++s) {
      const int kB = k0 + ((s < 8) ? (8 * half + s) : (16 + 8 * half + s - 8));
      bm[s] = fc1p[(size_t)kB * NP + n0 + lcol];
    }
    c = __builtin_amdgcn_wmma_f32_16x16x32_bf16(false, a, false, bm,
                                                (short)0, c, false, false);
  }
  {  // tail: K = 288..319 (guarded at 300; fc1p rows 300..319 are zero)
    const int k0 = 288;
    v16bf a = cat16(load8_f32_bf(erow, k0 + 8 * half, NE),
                    load8_f32_bf(erow, k0 + 16 + 8 * half, NE));
    v16bf bm;
#pragma unroll
    for (int s = 0; s < 16; ++s) {
      const int kB = k0 + ((s < 8) ? (8 * half + s) : (16 + 8 * half + s - 8));
      bm[s] = fc1p[(size_t)kB * NP + n0 + lcol];
    }
    c = __builtin_amdgcn_wmma_f32_16x16x32_bf16(false, a, false, bm,
                                                (short)0, c, false, false);
  }
  const float bias = fc1_bp[n0 + lcol];
#pragma unroll
  for (int r = 0; r < 8; ++r) {
    const int row = m0 + 8 * half + r;
    const float v = fmaxf(c[r] + bias, 0.f);
    Hx[(size_t)row * NP + n0 + lcol] = (__bf16)v;
  }
}

// ---------------------------------------------------------------------------
// Kernel 2: persistent RNN scan. 19 WGs (one 16-col slice each), weight slices
// in LDS (bf16), x_t double-buffered in LDS via async global->LDS prefetch,
// hidden ping-pong via L2-resident global buffers, device barrier 2x/step,
// per-wave register-resident running max.
// ---------------------------------------------------------------------------
__global__ __launch_bounds__(128) void rnn_scan_kernel(
    const __bf16* __restrict__ Hx, const __bf16* __restrict__ B1p,
    const __bf16* __restrict__ B2p, const float* __restrict__ l1_bp,
    const float* __restrict__ l2_bp, __bf16* H0, __bf16* H1,
    float* __restrict__ outsMax, int* cnt) {
  __shared__ __align__(16) __bf16 sb1[KC * 16];     // l1 weight slice [608][16]
  __shared__ __align__(16) __bf16 sb2[KE * 16];     // l2 weight slice [320][16]
  __shared__ __align__(16) __bf16 sx[2][NB * NP];   // x_t double buffer

  const int tid = threadIdx.x;
  const int wave = tid >> 5, lane = tid & 31;
  const int half = lane >> 4, lcol = lane & 15;
  const int n0 = blockIdx.x * 16;                  // owned output columns
  const int m0 = wave * 16;                        // owned batch rows
  const int am = m0 + lcol;

  for (int i = tid; i < KC * 16; i += 128) {
    const int k = i >> 4, n = i & 15;
    sb1[i] = B1p[(size_t)k * NP + n0 + n];
  }
  for (int i = tid; i < KE * 16; i += 128) {
    const int k = i >> 4, n = i & 15;
    sb2[i] = B2p[(size_t)k * NP + n0 + n];
  }
  // kick off async prefetch of x_0 into sx[0]
  const unsigned sx_base = (unsigned)(uintptr_t)(void*)&sx[0][0];
  const unsigned sx_stride = (unsigned)(NB * NP * 2);
  async_copy_xt(Hx, sx_base, tid);
  __syncthreads();

  const float bias1 = l1_bp[n0 + lcol];
  const float bias2 = l2_bp[n0 + lcol];
  v8f mx = vneginf8();
  int phase = 0;

  for (int t = 0; t < TT; ++t) {
    const __bf16* Hin  = (t & 1) ? H1 : H0;
    __bf16*       Hout = (t & 1) ? H0 : H1;
    ++phase; gbar(cnt, NWG * phase);               // Hin stable, WAR-safe

    // sx[t&1] holds x_t once our own async ops drain + block barrier
    wait_async0();
    __syncthreads();
    if (t + 1 < TT)                                 // prefetch x_{t+1}
      async_copy_xt(Hx + (size_t)(t + 1) * NB * NP,
                    sx_base + (unsigned)((t + 1) & 1) * sx_stride, tid);
    const __bf16* xt = &sx[t & 1][0];

    // ---- l1: hidden = sigmoid([x_t, h_{t-1}] @ l1_w^T + b1), K = 608
    v8f c = vzero8();
#pragma unroll 2
    for (int kk = 0; kk < KC / 32; ++kk) {
      const int k0 = kk * 32;
      const int kb0 = k0 + 8 * half;
      const int kb1 = kb0 + 16;
      const __bf16* p0 = (kb0 < NP) ? (xt + am * NP + kb0)
                                    : (Hin + (size_t)am * HS + (kb0 - NP));
      const __bf16* p1 = (kb1 < NP) ? (xt + am * NP + kb1)
                                    : (Hin + (size_t)am * HS + (kb1 - NP));
      v16bf a = cat16(*(const v8bf*)p0, *(const v8bf*)p1);
      v16bf bm;
#pragma unroll
      for (int s = 0; s < 16; ++s) {
        const int kB = k0 + ((s < 8) ? (8 * half + s) : (16 + 8 * half + s - 8));
        bm[s] = sb1[kB * 16 + lcol];
      }
      c = __builtin_amdgcn_wmma_f32_16x16x32_bf16(false, a, false, bm,
                                                  (short)0, c, false, false);
    }
#pragma unroll
    for (int r = 0; r < 8; ++r) {
      const float hv = 1.f / (1.f + __expf(-(c[r] + bias1)));
      const int row = m0 + 8 * half + r;
      Hout[(size_t)row * HS + n0 + lcol] = (__bf16)hv;
    }

    ++phase; gbar(cnt, NWG * phase);               // full new hidden visible

    // ---- l2: out = hidden @ l2_w^T + b2, K = 320 (padded)
    v8f c2 = vzero8();
#pragma unroll 2
    for (int kk = 0; kk < KE / 32; ++kk) {
      const int k0 = kk * 32;
      const int kb0 = k0 + 8 * half;
      v16bf a = cat16(*(const v8bf*)(Hout + (size_t)am * HS + kb0),
                      *(const v8bf*)(Hout + (size_t)am * HS + kb0 + 16));
      v16bf bm;
#pragma unroll
      for (int s = 0; s < 16; ++s) {
        const int kB = k0 + ((s < 8) ? (8 * half + s) : (16 + 8 * half + s - 8));
        bm[s] = sb2[kB * 16 + lcol];
      }
      c2 = __builtin_amdgcn_wmma_f32_16x16x32_bf16(false, a, false, bm,
                                                   (short)0, c2, false, false);
    }
#pragma unroll
    for (int r = 0; r < 8; ++r) mx[r] = fmaxf(mx[r], c2[r] + bias2);
  }

#pragma unroll
  for (int r = 0; r < 8; ++r) {
    const int row = m0 + 8 * half + r;
    outsMax[(size_t)row * NP + n0 + lcol] = mx[r];
  }
}

// ---------------------------------------------------------------------------
// Kernel 3: out[b][o] = pooled[b] . fc2_w[o] + fc2_b[o]   (f32, 128 dots)
// ---------------------------------------------------------------------------
__global__ void final_fc2_kernel(const float* __restrict__ outsMax,
                                 const float* __restrict__ fc2_w,
                                 const float* __restrict__ fc2_b,
                                 float* __restrict__ out) {
  const int tid = threadIdx.x;           // 128 = 64 batches * 2 outputs
  const int b = tid >> 1, o = tid & 1;
  float acc = fc2_b[o];
  for (int k = 0; k < NH; ++k)
    acc += outsMax[(size_t)b * NP + k] * fc2_w[o * NH + k];
  out[b * 2 + o] = acc;
}

// ---------------------------------------------------------------------------
extern "C" void kernel_launch(void* const* d_in, const int* in_sizes, int n_in,
                              void* d_out, int out_size, void* d_ws, size_t ws_size,
                              hipStream_t stream) {
  const int*   x     = (const int*)d_in[0];
  const float* emb   = (const float*)d_in[1];
  const float* fc1_w = (const float*)d_in[2];
  const float* fc1_b = (const float*)d_in[3];
  const float* l1_w  = (const float*)d_in[4];
  const float* l1_b  = (const float*)d_in[5];
  const float* l2_w  = (const float*)d_in[6];
  const float* l2_b  = (const float*)d_in[7];
  const float* fc2_w = (const float*)d_in[8];
  const float* fc2_b = (const float*)d_in[9];
  float* out = (float*)d_out;
  char* ws = (char*)d_ws;

  // workspace layout (256B aligned segments), ~20.9 MB total
  size_t off = 0;
  auto take = [&](size_t bytes) { size_t o = off; off = (off + bytes + 255) & ~(size_t)255; return o; };
  __bf16* fc1p   = (__bf16*)(ws + take((size_t)KE * NP * 2));
  __bf16* b1p    = (__bf16*)(ws + take((size_t)KC * NP * 2));
  __bf16* b2p    = (__bf16*)(ws + take((size_t)KE * NP * 2));
  float*  fc1_bp = (float*) (ws + take((size_t)NP * 4));
  float*  l1_bp  = (float*) (ws + take((size_t)NP * 4));
  float*  l2_bp  = (float*) (ws + take((size_t)NP * 4));
  __bf16* h0     = (__bf16*)(ws + take((size_t)NB * HS * 2));
  __bf16* h1     = (__bf16*)(ws + take((size_t)NB * HS * 2));
  float*  omax   = (float*) (ws + take((size_t)NB * NP * 4));
  int*    cnt    = (int*)   (ws + take(64));
  __bf16* Hx     = (__bf16*)(ws + take((size_t)TT * NB * NP * 2));

  const int pack_total = KE*NP + KC*NP + KE*NP + 3*NP + 2*NB*HS + 16;
  pack_kernel<<<(pack_total + 255) / 256, 256, 0, stream>>>(
      fc1_w, fc1_b, l1_w, l1_b, l2_w, l2_b,
      fc1p, b1p, b2p, fc1_bp, l1_bp, l2_bp, h0, h1, cnt);

  // (TT*NB/16) * (NP/16) tiles, 4 waves/block
  embed_fc1_kernel<<<(TT * NB / 16) * NWG / 4, 128, 0, stream>>>(
      x, emb, fc1p, fc1_bp, Hx);

  rnn_scan_kernel<<<NWG, 128, 0, stream>>>(
      Hx, b1p, b2p, l1_bp, l2_bp, h0, h1, omax, cnt);

  final_fc2_kernel<<<1, 128, 0, stream>>>(omax, fc2_w, fc2_b, out);
}
